// RelativeMultiHeadAttention_44049184588378
// MI455X (gfx1250) — compile-verified, matrix operand order checked by reading, not run on hardware
//
#include <hip/hip_runtime.h>
#include <hip/hip_bf16.h>

typedef __attribute__((ext_vector_type(16))) _Float16 v16h;
typedef __attribute__((ext_vector_type(8)))  _Float16 v8h;
typedef __attribute__((ext_vector_type(8)))  float    v8f;

typedef unsigned int u32x4  __attribute__((ext_vector_type(4)));
typedef int          i32x4v __attribute__((ext_vector_type(4)));
typedef int          i32x8v __attribute__((ext_vector_type(8)));

union Frag16 { v16h v; v8h hh[2]; };

#if defined(__has_builtin)
#  if __has_builtin(__builtin_amdgcn_tensor_load_to_lds) && \
      __has_builtin(__builtin_amdgcn_s_wait_tensorcnt)
#    define HAVE_TDM 1
#  endif
#endif
#ifndef HAVE_TDM
#  define HAVE_TDM 0
#endif

// ---------------------------------------------------------------------------
// TDM: async DMA of a 2D f16 tile (tile_d0 elems x tile_d1 rows) from global
// into LDS. Rows are 64 halfs (128B) and LDS rows are padded to 72 halfs:
// pad_interval code 4 (=16 x 8B = 128B), pad_amount code 3 (=4 DWORDs = 16B).
// Descriptor layout per CDNA5 ISA.
// ---------------------------------------------------------------------------
#if HAVE_TDM
__device__ __forceinline__ unsigned lds_off(const void* p) {
  return (unsigned)(unsigned long long)p;  // LDS aperture: low 32 bits = offset
}
__device__ __forceinline__ void tdm_load_tile_2d(const void* gptr,
                                                 unsigned lds_byte,
                                                 unsigned tile_d0,
                                                 unsigned tile_d1,
                                                 unsigned long long stride_el) {
  unsigned long long ga = (unsigned long long)gptr;
  u32x4 g0 = {1u,                                   // count=1, user mode
              lds_byte,                             // lds_addr
              (unsigned)ga,                         // global_addr[31:0]
              (unsigned)((ga >> 32) & 0x01FFFFFFu)  // global_addr[56:32]
                  | 0x80000000u};                   // type=2 ("image")
  unsigned td0 = (unsigned)stride_el;               // tensor_dim0 >= tile_d0
  unsigned td1 = 0x40000000u;                       // huge tensor_dim1 (no OOB)
  i32x8v g1;
  g1[0] = (int)((1u << 16)      // data_size = 2 bytes
                | (1u << 20)    // pad_enable
                | (4u << 22)    // pad_interval: 128B
                | (3u << 25));  // pad_amount: 4 DWORDs
  g1[1] = (int)((td0 & 0xFFFFu) << 16);                  // tensor_dim0[15:0]
  g1[2] = (int)((td0 >> 16) | ((td1 & 0xFFFFu) << 16));  // dim0 hi | dim1 lo
  g1[3] = (int)((td1 >> 16) | (tile_d0 << 16));          // dim1 hi | tile_dim0
  g1[4] = (int)tile_d1;                                  // tile_dim1, tile_dim2=0
  g1[5] = (int)(unsigned)stride_el;                      // dim0_stride[31:0]
  g1[6] = (int)(unsigned)(stride_el >> 32);              // dim0_stride[47:32]
  g1[7] = 0;
  i32x4v z4 = {0, 0, 0, 0};
#if __clang_major__ >= 23
  i32x8v z8 = {0, 0, 0, 0, 0, 0, 0, 0};
  __builtin_amdgcn_tensor_load_to_lds(g0, g1, z4, z4, z8, 0);
#else
  __builtin_amdgcn_tensor_load_to_lds(g0, g1, z4, z4, 0);
#endif
}
#endif  // HAVE_TDM

// ---------------------------------------------------------------------------
// DPP16 row butterflies: reduce across the 16 lanes of each wave half.
// ---------------------------------------------------------------------------
template <int CTRL>
__device__ __forceinline__ float dpp_xfer(float x) {
  int xi = __builtin_bit_cast(int, x);
  int r  = __builtin_amdgcn_update_dpp(xi, xi, CTRL, 0xF, 0xF, false);
  return __builtin_bit_cast(float, r);
}
__device__ __forceinline__ float row16_max(float v) {
  v = fmaxf(v, dpp_xfer<0xB1>(v));   // quad_perm(1,0,3,2)
  v = fmaxf(v, dpp_xfer<0x4E>(v));   // quad_perm(2,3,0,1)
  v = fmaxf(v, dpp_xfer<0x141>(v));  // row_half_mirror
  v = fmaxf(v, dpp_xfer<0x140>(v));  // row_mirror
  return v;
}
__device__ __forceinline__ float row16_sum(float v) {
  v += dpp_xfer<0xB1>(v);
  v += dpp_xfer<0x4E>(v);
  v += dpp_xfer<0x141>(v);
  v += dpp_xfer<0x140>(v);
  return v;
}

// ---------------------------------------------------------------------------
// f32 -> f16 conversion (grid-stride)
// ---------------------------------------------------------------------------
__global__ void cvt_f16_kernel(const float* __restrict__ in,
                               _Float16* __restrict__ out, long long n) {
  long long i = (long long)blockIdx.x * blockDim.x + threadIdx.x;
  long long stride = (long long)gridDim.x * blockDim.x;
  for (; i < n; i += stride) out[i] = (_Float16)in[i];
}

// ---------------------------------------------------------------------------
// In-place RoPE on [B, T, nheads*64] f16; `scale` folds 1/sqrt(dh) into Q.
// ---------------------------------------------------------------------------
__global__ void rope_inplace_kernel(_Float16* __restrict__ x,
                                    int B, int T, int nheads, float scale) {
  int idx = blockIdx.x * blockDim.x + threadIdx.x;
  int total = B * T * nheads * 32;
  if (idx >= total) return;
  int d   = idx & 31;
  int row = idx >> 5;
  int t   = (row / nheads) % T;
  _Float16* p = x + (size_t)row * 64;
  float inv = __powf(10000.0f, -(float)d * (1.0f / 32.0f));
  float ang = (float)t * inv;
  float c = __cosf(ang), s = __sinf(ang);
  float x0 = (float)p[d], x1 = (float)p[d + 32];
  p[d]      = (_Float16)((x0 * c - x1 * s) * scale);
  p[d + 32] = (_Float16)((x1 * c + x0 * s) * scale);
}

// ---------------------------------------------------------------------------
// WMMA GEMM: C[M,N] = A[M,K] * W[N,K]^T + bias[N]
// Block tile 128x64, 8 waves, each wave 2x2 WMMA tiles. K staged in 64-col
// chunks, double-buffered LDS filled by the TDM (wave 0 issues, all waves
// compute) with one-chunk lookahead.
// ---------------------------------------------------------------------------
template <bool OUT_F16>
__global__ void gemm_wmma_kernel(const _Float16* __restrict__ A,
                                 const _Float16* __restrict__ W,
                                 const float* __restrict__ bias,
                                 void* __restrict__ Cout,
                                 int M, int N, int K) {
  __shared__ __align__(16) _Float16 As[2][128][72];
  __shared__ __align__(16) _Float16 Bs[2][64][72];

  const int tid  = threadIdx.x;
  const int wave = tid >> 5;
  const int lane = tid & 31;
  const int hx   = lane >> 4;
  const int l16  = lane & 15;
  const int wm   = wave & 3;
  const int wn   = wave >> 2;
  const int m0   = blockIdx.y * 128;
  const int n0   = blockIdx.x * 64;

  v8f acc[2][2] = {};

#if HAVE_TDM
  if (wave == 0) {
    tdm_load_tile_2d(A + (size_t)m0 * K, lds_off(&As[0][0][0]), 64, 128, K);
    tdm_load_tile_2d(W + (size_t)n0 * K, lds_off(&Bs[0][0][0]), 64, 64, K);
  }
#else
  {
    int r = tid >> 1, cb = (tid & 1) * 32;
    const _Float16* g = A + (size_t)(m0 + r) * K + cb;
#pragma unroll
    for (int i = 0; i < 4; ++i)
      *(v8h*)&As[0][r][cb + 8 * i] = *(const v8h*)(g + 8 * i);
  }
  {
    int r = tid >> 2, cb = (tid & 3) * 16;
    const _Float16* g = W + (size_t)(n0 + r) * K + cb;
#pragma unroll
    for (int i = 0; i < 2; ++i)
      *(v8h*)&Bs[0][r][cb + 8 * i] = *(const v8h*)(g + 8 * i);
  }
#endif

  for (int k0 = 0; k0 < K; k0 += 64) {
    const int cur = (k0 >> 6) & 1;
    const int nxt = cur ^ 1;
#if HAVE_TDM
    if (wave == 0) {
      if (k0 + 64 < K) {
        tdm_load_tile_2d(A + (size_t)m0 * K + k0 + 64,
                         lds_off(&As[nxt][0][0]), 64, 128, K);
        tdm_load_tile_2d(W + (size_t)n0 * K + k0 + 64,
                         lds_off(&Bs[nxt][0][0]), 64, 64, K);
        __builtin_amdgcn_s_wait_tensorcnt(2);  // current tile's 2 DMAs done
      } else {
        __builtin_amdgcn_s_wait_tensorcnt(0);
      }
    }
#else
    if (k0 + 64 < K) {
      {
        int r = tid >> 1, cb = (tid & 1) * 32;
        const _Float16* g = A + (size_t)(m0 + r) * K + k0 + 64 + cb;
#pragma unroll
        for (int i = 0; i < 4; ++i)
          *(v8h*)&As[nxt][r][cb + 8 * i] = *(const v8h*)(g + 8 * i);
      }
      {
        int r = tid >> 2, cb = (tid & 3) * 16;
        const _Float16* g = W + (size_t)(n0 + r) * K + k0 + 64 + cb;
#pragma unroll
        for (int i = 0; i < 2; ++i)
          *(v8h*)&Bs[nxt][r][cb + 8 * i] = *(const v8h*)(g + 8 * i);
      }
    }
#endif
    __syncthreads();  // current tile visible to all waves

#pragma unroll
    for (int kk = 0; kk < 64; kk += 32) {
      Frag16 a[2], b[2];
#pragma unroll
      for (int i = 0; i < 2; ++i) {
        int row = wm * 32 + i * 16 + l16;
        a[i].hh[0] = *(const v8h*)&As[cur][row][kk + 8 * hx];
        a[i].hh[1] = *(const v8h*)&As[cur][row][kk + 16 + 8 * hx];
      }
#pragma unroll
      for (int j = 0; j < 2; ++j) {
        int row = wn * 32 + j * 16 + l16;
        b[j].hh[0] = *(const v8h*)&Bs[cur][row][kk + 8 * hx];
        b[j].hh[1] = *(const v8h*)&Bs[cur][row][kk + 16 + 8 * hx];
      }
#pragma unroll
      for (int i = 0; i < 2; ++i)
#pragma unroll
        for (int j = 0; j < 2; ++j)
          acc[i][j] = __builtin_amdgcn_wmma_f32_16x16x32_f16(
              false, a[i].v, false, b[j].v, (short)0, acc[i][j], false, false);
    }
    __syncthreads();  // all waves done with current before it is overwritten
  }

#pragma unroll
  for (int i = 0; i < 2; ++i)
#pragma unroll
    for (int j = 0; j < 2; ++j) {
      int col = n0 + wn * 32 + j * 16 + l16;
      float bv = bias[col];
#pragma unroll
      for (int r = 0; r < 8; ++r) {
        int row = m0 + wm * 32 + i * 16 + r + 8 * hx;
        float val = acc[i][j][r] + bv;
        if (OUT_F16)
          ((_Float16*)Cout)[(size_t)row * N + col] = (_Float16)val;
        else
          ((float*)Cout)[(size_t)row * N + col] = val;
      }
    }
}

// ---------------------------------------------------------------------------
// Flash MQA attention (single KV head shared across H=16 query heads).
// grid.x = B*H, grid.y = T/128. 8 waves per block; wave w owns 16 query rows.
// K tiles: double-buffered LDS via TDM (one-block lookahead). V tiles:
// cooperative transposed staging. Row stats: DPP16 butterflies.
// ---------------------------------------------------------------------------
__global__ void mqa_flash_kernel(const _Float16* __restrict__ Q,  // [B,T,H*64]
                                 const _Float16* __restrict__ Kk, // [B,T,64]
                                 const _Float16* __restrict__ V,  // [B,T,64]
                                 _Float16* __restrict__ ctx,      // [B,T,H*64]
                                 int B, int T, int H) {
  __shared__ __align__(16) _Float16 Ks[2][32][72];   // K tiles, double buffer
  __shared__ __align__(16) _Float16 VT[64][48];      // V^T tile
  __shared__ __align__(16) _Float16 Pls[8][16][40];  // per-wave P staging

  const int tid  = threadIdx.x;
  const int wave = tid >> 5;
  const int lane = tid & 31;
  const int hx   = lane >> 4;
  const int l16  = lane & 15;
  const int bh   = blockIdx.x;
  const int b    = bh / H;
  const int h    = bh % H;
  const int q0   = blockIdx.y * 128 + wave * 16;
  const size_t qstride = (size_t)H * 64;

  // Q fragments (16 rows x 64 dims); 1/sqrt(dh) pre-folded by RoPE kernel.
  Frag16 qa[2];
  {
    const _Float16* qp = Q + ((size_t)b * T + q0 + l16) * qstride + h * 64;
#pragma unroll
    for (int ks = 0; ks < 2; ++ks) {
      qa[ks].hh[0] = *(const v8h*)(qp + 32 * ks + 8 * hx);
      qa[ks].hh[1] = *(const v8h*)(qp + 32 * ks + 16 + 8 * hx);
    }
  }

  v8f acc[4] = {};
  float mrow[8], lrow[8];
#pragma unroll
  for (int r = 0; r < 8; ++r) { mrow[r] = -1e30f; lrow[r] = 0.0f; }

#if HAVE_TDM
  if (wave == 0)
    tdm_load_tile_2d(Kk + (size_t)b * T * 64, lds_off(&Ks[0][0][0]), 64, 32, 64);
#endif

  for (int s0 = 0; s0 < T; s0 += 32) {
    const int cur = (s0 >> 5) & 1;
    const int nxt = cur ^ 1;
    __syncthreads();  // all waves done reading previous tiles
    // V tile: cooperative transposed stage.
    {
      int key = tid >> 3;
      int cb  = (tid & 7) * 8;
      v8h vv = *(const v8h*)(V + ((size_t)b * T + s0 + key) * 64 + cb);
#pragma unroll
      for (int e = 0; e < 8; ++e) VT[cb + e][key] = vv[e];
    }
#if HAVE_TDM
    if (wave == 0) {
      if (s0 + 32 < T) {
        tdm_load_tile_2d(Kk + ((size_t)b * T + s0 + 32) * 64,
                         lds_off(&Ks[nxt][0][0]), 64, 32, 64);
        __builtin_amdgcn_s_wait_tensorcnt(1);  // current K tile done
      } else {
        __builtin_amdgcn_s_wait_tensorcnt(0);
      }
    }
#else
    {
      int key = tid >> 3;
      int cb  = (tid & 7) * 8;
      *(v8h*)&Ks[cur][key][cb] =
          *(const v8h*)(Kk + ((size_t)b * T + s0 + key) * 64 + cb);
    }
#endif
    __syncthreads();

    // S = Q * K^T for two 16-key n-tiles (K fragments from LDS).
    v8f Sf[2];
#pragma unroll
    for (int j = 0; j < 2; ++j) {
      Frag16 kf[2];
#pragma unroll
      for (int ks = 0; ks < 2; ++ks) {
        kf[ks].hh[0] = *(const v8h*)&Ks[cur][16 * j + l16][32 * ks + 8 * hx];
        kf[ks].hh[1] = *(const v8h*)&Ks[cur][16 * j + l16][32 * ks + 16 + 8 * hx];
      }
      v8f c = {};
      c = __builtin_amdgcn_wmma_f32_16x16x32_f16(false, qa[0].v, false, kf[0].v,
                                                 (short)0, c, false, false);
      c = __builtin_amdgcn_wmma_f32_16x16x32_f16(false, qa[1].v, false, kf[1].v,
                                                 (short)0, c, false, false);
      Sf[j] = c;
    }

    // Online softmax; row M = r + 8*hx lives on the 16 lanes of half hx.
#pragma unroll
    for (int r = 0; r < 8; ++r) {
      float sv0 = Sf[0][r];
      float sv1 = Sf[1][r];
      float mx = row16_max(fmaxf(sv0, sv1));
      float mnew  = fmaxf(mrow[r], mx);
      float alpha = __expf(mrow[r] - mnew);
      mrow[r] = mnew;
      float p0 = __expf(sv0 - mnew);
      float p1 = __expf(sv1 - mnew);
      Sf[0][r] = p0;
      Sf[1][r] = p1;
      lrow[r] = lrow[r] * alpha + row16_sum(p0 + p1);
#pragma unroll
      for (int n = 0; n < 4; ++n) acc[n][r] = acc[n][r] * alpha;
    }

    // Re-materialize P (16x32) in A-fragment layout via wave-private LDS.
#pragma unroll
    for (int j = 0; j < 2; ++j)
#pragma unroll
      for (int r = 0; r < 8; ++r)
        Pls[wave][r + 8 * hx][16 * j + l16] = (_Float16)Sf[j][r];
    __builtin_amdgcn_wave_barrier();

    Frag16 pf;
    pf.hh[0] = *(const v8h*)&Pls[wave][l16][8 * hx];
    pf.hh[1] = *(const v8h*)&Pls[wave][l16][16 + 8 * hx];

    // acc += P * V for four 16-dim n-tiles.
#pragma unroll
    for (int n = 0; n < 4; ++n) {
      Frag16 vf;
      const int dim = n * 16 + l16;
      vf.hh[0] = *(const v8h*)&VT[dim][8 * hx];
      vf.hh[1] = *(const v8h*)&VT[dim][16 + 8 * hx];
      acc[n] = __builtin_amdgcn_wmma_f32_16x16x32_f16(
          false, pf.v, false, vf.v, (short)0, acc[n], false, false);
    }
  }

  // Normalize and write ctx in [B,T,H*64] layout (ready for O-projection).
#pragma unroll
  for (int n = 0; n < 4; ++n)
#pragma unroll
    for (int r = 0; r < 8; ++r) {
      int trow = q0 + r + 8 * hx;
      ctx[((size_t)b * T + trow) * qstride + h * 64 + n * 16 + l16] =
          (_Float16)(acc[n][r] / lrow[r]);
    }
}

// ---------------------------------------------------------------------------
// Launcher
// ---------------------------------------------------------------------------
extern "C" void kernel_launch(void* const* d_in, const int* in_sizes, int n_in,
                              void* d_out, int out_size, void* d_ws,
                              size_t ws_size, hipStream_t stream) {
  const float* x  = (const float*)d_in[0];
  const float* xk = (const float*)d_in[1];
  const float* xv = (const float*)d_in[2];
  const float* Wq = (const float*)d_in[3];
  const float* bq = (const float*)d_in[4];
  const float* Wk = (const float*)d_in[5];
  const float* bk = (const float*)d_in[6];
  const float* Wv = (const float*)d_in[7];
  const float* bv = (const float*)d_in[8];
  const float* Wo = (const float*)d_in[9];
  const float* bo = (const float*)d_in[10];

  const int D = 1024, H = 16, dh = 64, T = 2048;
  const long long BT = (long long)in_sizes[0] / D;
  const int B = (int)(BT / T);

  char* w = (char*)d_ws;
  _Float16* x16  = (_Float16*)w; w += (size_t)BT * D * 2;
  _Float16* xk16 = (_Float16*)w; w += (size_t)BT * D * 2;
  _Float16* xv16 = (_Float16*)w; w += (size_t)BT * D * 2;
  _Float16* wq16 = (_Float16*)w; w += (size_t)D * D * 2;
  _Float16* wk16 = (_Float16*)w; w += (size_t)dh * D * 2;
  _Float16* wv16 = (_Float16*)w; w += (size_t)dh * D * 2;
  _Float16* wo16 = (_Float16*)w; w += (size_t)D * D * 2;
  _Float16* q16  = (_Float16*)w; w += (size_t)BT * D * 2;
  _Float16* k16  = (_Float16*)w; w += (size_t)BT * dh * 2;
  _Float16* v16  = (_Float16*)w; w += (size_t)BT * dh * 2;
  _Float16* c16  = (_Float16*)w; w += (size_t)BT * D * 2;

  auto cvt = [&](const float* in, _Float16* out, long long n) {
    int blocks = (int)((n + 255) / 256);
    if (blocks > 65535) blocks = 65535;
    cvt_f16_kernel<<<blocks, 256, 0, stream>>>(in, out, n);
  };
  cvt(x,  x16,  BT * D);
  cvt(xk, xk16, BT * D);
  cvt(xv, xv16, BT * D);
  cvt(Wq, wq16, (long long)D * D);
  cvt(Wk, wk16, (long long)dh * D);
  cvt(Wv, wv16, (long long)dh * D);
  cvt(Wo, wo16, (long long)D * D);

  dim3 blk(256);
  // Projections
  gemm_wmma_kernel<true><<<dim3(D / 64, (int)(BT / 128)), blk, 0, stream>>>(
      x16, wq16, bq, (void*)q16, (int)BT, D, D);
  gemm_wmma_kernel<true><<<dim3(1, (int)(BT / 128)), blk, 0, stream>>>(
      xk16, wk16, bk, (void*)k16, (int)BT, dh, D);
  gemm_wmma_kernel<true><<<dim3(1, (int)(BT / 128)), blk, 0, stream>>>(
      xv16, wv16, bv, (void*)v16, (int)BT, dh, D);

  // RoPE (in-place, pair-per-thread); fold 1/sqrt(dh) into Q.
  rope_inplace_kernel<<<(B * T * H * 32) / 256, 256, 0, stream>>>(
      q16, B, T, H, 0.125f);
  rope_inplace_kernel<<<(B * T * 1 * 32) / 256, 256, 0, stream>>>(
      k16, B, T, 1, 1.0f);

  // Flash MQA attention
  mqa_flash_kernel<<<dim3(B * H, T / 128), blk, 0, stream>>>(
      q16, k16, v16, c16, B, T, H);

  // Output projection (f32 out)
  gemm_wmma_kernel<false><<<dim3(D / 64, (int)(BT / 128)), blk, 0, stream>>>(
      c16, wo16, bo, d_out, (int)BT, D, D);
}
